// SingleSelfAttention_60370060313342
// MI455X (gfx1250) — matrix operand order repaired and checked by reading, not compiled, not run on hardware
//
#include <hip/hip_runtime.h>
#include <hip/hip_bf16.h>

typedef __bf16 bf16;
typedef __attribute__((ext_vector_type(16))) __bf16 v16bf;
typedef __attribute__((ext_vector_type(8)))  float  v8f;
typedef __attribute__((ext_vector_type(4)))  int    v4i;

#define BATCH 8
#define SEQ   2048
#define EMB   1024
#define NDIM  1024

// LDS tile: 128 rows x 32 k-elements, padded to 40 (80B pitch = 5x16B, keeps
// b128 alignment and spreads banks: dword-bank stride 20, gcd(20,64)=4 -> 16
// distinct banks across the 16 lanes of a fragment read).
#define LDS_STRIDE 40
#define TILE_ELEMS (128 * LDS_STRIDE)

#if defined(__HIP_DEVICE_COMPILE__) && \
    __has_builtin(__builtin_amdgcn_global_load_async_to_lds_b128) && \
    __has_builtin(__builtin_amdgcn_s_wait_asynccnt)
#define USE_ASYNC 1
typedef __attribute__((address_space(1))) v4i g_v4i;   // global
typedef __attribute__((address_space(3))) v4i l_v4i;   // LDS
#else
#define USE_ASYNC 0
#endif

union FragU { v16bf v; v4i d[2]; };

// ---- WMMA fragment loaders (wave32, bf16 16x16x32) ----
// A: 16x32 tile from row-major [M x K] memory (ld in elements).
__device__ __forceinline__ v16bf load_a_frag(const bf16* __restrict__ base, int ld) {
    int lane = threadIdx.x & 31;
    int m    = lane & 15;
    int koff = (lane >> 4) << 3;          // 0 or 8
    const bf16* p = base + (size_t)m * ld + koff;
    FragU u;
    u.d[0] = *(const v4i*)(p);            // K = koff .. koff+7
    u.d[1] = *(const v4i*)(p + 16);       // K = koff+16 .. koff+23
    return u.v;
}

// B: 32x16 tile from N-major (K contiguous) memory, i.e. Bt[n][k].
__device__ __forceinline__ v16bf load_b_frag(const bf16* __restrict__ base, int ld) {
    int lane  = threadIdx.x & 31;
    int n     = lane & 15;
    int kbase = (lane >> 4) << 3;         // 0 or 8
    const bf16* p = base + (size_t)n * ld + kbase;
    FragU u;
    u.d[0] = *(const v4i*)(p);            // K = kbase .. kbase+7
    u.d[1] = *(const v4i*)(p + 16);       // K = kbase+16 .. kbase+23
    return u.v;
}

// Stage one 128x32 A tile and one 128x32 B tile into LDS.
// 512 16B-chunks per tile; 256 threads -> 2 chunks each per tile,
// 4 async instructions per wave per fill (matches s_wait_asynccnt 4).
__device__ __forceinline__ void fill_tiles(const bf16* __restrict__ gA, int lda,
                                           const bf16* __restrict__ gB, int ldb,
                                           bf16* ldsA, bf16* ldsB, int k0) {
    int t = threadIdx.x;
#pragma unroll
    for (int h = 0; h < 2; ++h) {
        int c   = t + h * 256;            // chunk id 0..511
        int row = c >> 2;
        int sub = (c & 3) * 8;            // element offset within the 32-wide row
        const bf16* ga = gA + (size_t)row * lda + k0 + sub;
        const bf16* gb = gB + (size_t)row * ldb + k0 + sub;
        bf16* la = ldsA + row * LDS_STRIDE + sub;
        bf16* lb = ldsB + row * LDS_STRIDE + sub;
#if USE_ASYNC
        __builtin_amdgcn_global_load_async_to_lds_b128((g_v4i*)ga, (l_v4i*)la, 0, 0);
        __builtin_amdgcn_global_load_async_to_lds_b128((g_v4i*)gb, (l_v4i*)lb, 0, 0);
#else
        *(v4i*)la = *(const v4i*)ga;
        *(v4i*)lb = *(const v4i*)gb;
#endif
    }
}

// Workgroup (256 thr = 8 waves) computes a 128x128 tile; wave w does the
// 64x32 sub-tile at (wm, wn). Double-buffered LDS staging of A/B k-slabs.
__device__ __forceinline__ void block_gemm(const bf16* __restrict__ gA, int lda,
                                           const bf16* __restrict__ gB, int ldb,
                                           int kdim,
                                           bf16* ldsA, bf16* ldsB,  // each [2*TILE_ELEMS]
                                           v8f (&acc)[4][2]) {
    int w  = threadIdx.x >> 5;
    int wm = (w >> 2) * 64;               // 0 or 64
    int wn = (w & 3) * 32;                // 0,32,64,96
    int nk = kdim >> 5;

    fill_tiles(gA, lda, gB, ldb, ldsA, ldsB, 0);
    for (int kk = 0; kk < nk; ++kk) {
        int cur = kk & 1;
        if (kk + 1 < nk)
            fill_tiles(gA, lda, gB, ldb,
                       ldsA + (cur ^ 1) * TILE_ELEMS, ldsB + (cur ^ 1) * TILE_ELEMS,
                       (kk + 1) * 32);
#if USE_ASYNC
        if (kk + 1 < nk) __builtin_amdgcn_s_wait_asynccnt(4);  // current slab landed
        else             __builtin_amdgcn_s_wait_asynccnt(0);
#endif
        __syncthreads();                  // slab visible to all waves

        const bf16* la = ldsA + cur * TILE_ELEMS + wm * LDS_STRIDE;
        const bf16* lb = ldsB + cur * TILE_ELEMS + wn * LDS_STRIDE;
        v16bf a[4], b[2];
#pragma unroll
        for (int i = 0; i < 4; ++i) a[i] = load_a_frag(la + i * 16 * LDS_STRIDE, LDS_STRIDE);
#pragma unroll
        for (int j = 0; j < 2; ++j) b[j] = load_b_frag(lb + j * 16 * LDS_STRIDE, LDS_STRIDE);
#pragma unroll
        for (int i = 0; i < 4; ++i)
#pragma unroll
            for (int j = 0; j < 2; ++j)
                acc[i][j] = __builtin_amdgcn_wmma_f32_16x16x32_bf16(
                    false, a[i], false, b[j], (short)0, acc[i][j], false, false);
        __syncthreads();                  // done reading before slab reuse
    }
}

// ---- elementwise helpers ----
__global__ void cvt_kernel(const float* __restrict__ src, bf16* __restrict__ dst, int n) {
    int i = blockIdx.x * blockDim.x + threadIdx.x;
    if (i < n) dst[i] = (bf16)src[i];
}

// W [EMB][NDIM] f32 -> WT [NDIM][EMB] bf16 (N-major, K contiguous)
__global__ void wtrans_kernel(const float* __restrict__ W, bf16* __restrict__ WT) {
    int idx = blockIdx.x * blockDim.x + threadIdx.x;
    if (idx >= EMB * NDIM) return;
    int n = idx & (NDIM - 1);
    int k = idx >> 10;
    WT[(size_t)n * EMB + k] = (bf16)W[idx];
}

// ---- QKV projection ----
__global__ __launch_bounds__(256, 1)
void qkv_gemm_kernel(const bf16* __restrict__ Xb, const bf16* __restrict__ WT,
                     bf16* __restrict__ Q, bf16* __restrict__ K, bf16* __restrict__ V) {
    __shared__ bf16 shA[2 * TILE_ELEMS];
    __shared__ bf16 shB[2 * TILE_ELEMS];
    int n0 = blockIdx.x * 128;
    int m0 = blockIdx.y * 128;
    int wsel = blockIdx.z;
    const bf16* Wt = WT + (size_t)wsel * NDIM * EMB;
    bf16* out = (wsel == 0) ? Q : (wsel == 1) ? K : V;

    v8f acc[4][2] = {};
    block_gemm(Xb + (size_t)m0 * EMB, EMB, Wt + (size_t)n0 * EMB, EMB, EMB, shA, shB, acc);

    int w  = threadIdx.x >> 5;
    int wm = (w >> 2) * 64, wn = (w & 3) * 32;
    int lane = threadIdx.x & 31;
    int n = lane & 15, mo = (lane >> 4) * 8;
#pragma unroll
    for (int i = 0; i < 4; ++i)
#pragma unroll
        for (int j = 0; j < 2; ++j)
#pragma unroll
            for (int r = 0; r < 8; ++r)
                out[(size_t)(m0 + wm + i * 16 + mo + r) * NDIM + (n0 + wn + j * 16 + n)] =
                    (bf16)acc[i][j][r];
}

// ---- scores: S = scale * Q K^T with causal mask ----
__global__ __launch_bounds__(256, 1)
void scores_kernel(const bf16* __restrict__ Qb, const bf16* __restrict__ Kb,
                   float* __restrict__ S) {
    int bsi = blockIdx.x, bti = blockIdx.y, b = blockIdx.z;
    if (bti < bsi) return;                // 128x128 block fully above diagonal
    __shared__ bf16 shA[2 * TILE_ELEMS];
    __shared__ bf16 shB[2 * TILE_ELEMS];
    const bf16* Qp = Qb + (size_t)b * SEQ * NDIM;
    const bf16* Kp = Kb + (size_t)b * SEQ * NDIM;
    float* Sp = S + (size_t)b * SEQ * SEQ;
    int t0 = bti * 128, s0 = bsi * 128;

    v8f acc[4][2] = {};
    block_gemm(Qp + (size_t)t0 * NDIM, NDIM, Kp + (size_t)s0 * NDIM, NDIM, NDIM, shA, shB, acc);

    const float scale = 0.03125f;         // 1024^-0.5
    int w  = threadIdx.x >> 5;
    int wm = (w >> 2) * 64, wn = (w & 3) * 32;
    int lane = threadIdx.x & 31;
    int n = lane & 15, mo = (lane >> 4) * 8;
#pragma unroll
    for (int i = 0; i < 4; ++i)
#pragma unroll
        for (int j = 0; j < 2; ++j)
#pragma unroll
            for (int r = 0; r < 8; ++r) {
                int t = t0 + wm + i * 16 + mo + r;
                int s = s0 + wn + j * 16 + n;
                Sp[(size_t)t * SEQ + s] =
                    (t >= s) ? acc[i][j][r] * scale : -__builtin_inff();
            }
}

// ---- query-axis softmax stats per column s ----
__global__ void colreduce_kernel(const float* __restrict__ S,
                                 float* __restrict__ M, float* __restrict__ Dinv) {
    int s = blockIdx.x * blockDim.x + threadIdx.x;
    int b = blockIdx.y;
    const float* Sp = S + (size_t)b * SEQ * SEQ;
    float m = -__builtin_inff(), d = 0.f;
    for (int t = s; t < SEQ; ++t) {
        float v = Sp[(size_t)t * SEQ + s];
        if (v > m) { d = d * __expf(m - v) + 1.f; m = v; }
        else       { d += __expf(v - m); }
    }
    M[b * SEQ + s]    = m;
    Dinv[b * SEQ + s] = 1.f / d;
}

// ---- P = exp(S - M[s]) (0 above diagonal) as bf16 ----
__global__ void probs_kernel(const float* __restrict__ S, const float* __restrict__ M,
                             bf16* __restrict__ P) {
    size_t idx   = (size_t)blockIdx.x * blockDim.x + threadIdx.x;
    size_t total = (size_t)BATCH * SEQ * SEQ;
    size_t step  = (size_t)gridDim.x * blockDim.x;
    for (; idx < total; idx += step) {
        int s = (int)(idx & (SEQ - 1));
        size_t r = idx >> 11;
        int t = (int)(r & (SEQ - 1));
        int b = (int)(r >> 11);
        float p = 0.f;
        if (t >= s) p = __expf(S[idx] - M[b * SEQ + s]);
        P[idx] = (bf16)p;
    }
}

// ---- V'[d][s] = V[s][d] * Dinv[s] ----
__global__ void vprep_kernel(const bf16* __restrict__ V, const float* __restrict__ Dinv,
                             bf16* __restrict__ VT) {
    size_t idx = (size_t)blockIdx.x * blockDim.x + threadIdx.x;
    if (idx >= (size_t)BATCH * SEQ * NDIM) return;
    int d = (int)(idx & (NDIM - 1));
    size_t r = idx >> 10;
    int s = (int)(r & (SEQ - 1));
    int b = (int)(r >> 11);
    float v = (float)V[idx] * Dinv[b * SEQ + s];
    VT[((size_t)b * NDIM + d) * SEQ + s] = (bf16)v;
}

// ---- out[b,t,d] = sum_{s<=t} P[t,s] * V'[d][s] ----
__global__ __launch_bounds__(256, 1)
void out_gemm_kernel(const bf16* __restrict__ P, const bf16* __restrict__ VT,
                     float* __restrict__ Out) {
    __shared__ bf16 shA[2 * TILE_ELEMS];
    __shared__ bf16 shB[2 * TILE_ELEMS];
    int d0 = blockIdx.x * 128;
    int t0 = blockIdx.y * 128;
    int b  = blockIdx.z;
    const bf16* Pp = P  + (size_t)b * SEQ * SEQ;
    const bf16* Vp = VT + (size_t)b * NDIM * SEQ;
    float* Op = Out + (size_t)b * SEQ * NDIM;

    v8f acc[4][2] = {};
    int kmax = t0 + 128;                  // triangular contraction bound
    block_gemm(Pp + (size_t)t0 * SEQ, SEQ, Vp + (size_t)d0 * SEQ, SEQ, kmax, shA, shB, acc);

    int w  = threadIdx.x >> 5;
    int wm = (w >> 2) * 64, wn = (w & 3) * 32;
    int lane = threadIdx.x & 31;
    int n = lane & 15, mo = (lane >> 4) * 8;
#pragma unroll
    for (int i = 0; i < 4; ++i)
#pragma unroll
        for (int j = 0; j < 2; ++j)
#pragma unroll
            for (int r = 0; r < 8; ++r)
                Op[(size_t)(t0 + wm + i * 16 + mo + r) * NDIM + (d0 + wn + j * 16 + n)] =
                    acc[i][j][r];
}

extern "C" void kernel_launch(void* const* d_in, const int* in_sizes, int n_in,
                              void* d_out, int out_size, void* d_ws, size_t ws_size,
                              hipStream_t stream) {
    const float* X  = (const float*)d_in[0];
    const float* Wq = (const float*)d_in[1];
    const float* Wk = (const float*)d_in[2];
    const float* Wv = (const float*)d_in[3];
    float* Out = (float*)d_out;
    char* ws = (char*)d_ws;

    // workspace layout (bytes)
    bf16*  Xb = (bf16*)(ws + 0);             // 33554432
    bf16*  WT = (bf16*)(ws + 33554432);      //  6291456
    bf16*  Qb = (bf16*)(ws + 39845888);      // 33554432
    bf16*  Kb = (bf16*)(ws + 73400320);      // 33554432
    bf16*  Vb = (bf16*)(ws + 106954752);     // 33554432
    float* S  = (float*)(ws + 140509184);    // 134217728
    float* Mx = (float*)(ws + 274726912);    //    65536
    float* Di = (float*)(ws + 274792448);    //    65536
    bf16*  P  = (bf16*)(ws + 274857984);     // 67108864
    bf16*  VT = (bf16*)(ws + 341966848);     // 33554432 (total ~375.5 MB)

    const int NX = BATCH * SEQ * EMB;        // 16777216
    const int NW = EMB * NDIM;               // 1048576

    cvt_kernel<<<(NX + 255) / 256, 256, 0, stream>>>(X, Xb, NX);
    wtrans_kernel<<<(NW + 255) / 256, 256, 0, stream>>>(Wq, WT);
    wtrans_kernel<<<(NW + 255) / 256, 256, 0, stream>>>(Wk, WT + (size_t)NW);
    wtrans_kernel<<<(NW + 255) / 256, 256, 0, stream>>>(Wv, WT + (size_t)2 * NW);

    qkv_gemm_kernel<<<dim3(NDIM / 128, BATCH * SEQ / 128, 3), 256, 0, stream>>>(Xb, WT, Qb, Kb, Vb);
    scores_kernel<<<dim3(SEQ / 128, SEQ / 128, BATCH), 256, 0, stream>>>(Qb, Kb, S);
    colreduce_kernel<<<dim3(SEQ / 256, BATCH), 256, 0, stream>>>(S, Mx, Di);
    probs_kernel<<<8192, 256, 0, stream>>>(S, Mx, P);
    vprep_kernel<<<(NX + 255) / 256, 256, 0, stream>>>(Vb, Di, VT);
    out_gemm_kernel<<<dim3(NDIM / 128, SEQ / 128, BATCH), 256, 0, stream>>>(P, VT, Out);
}
